// Lattice_Transformer_SeqLabel_50087908606770
// MI455X (gfx1250) — matrix operand order repaired and checked by reading, not compiled, not run on hardware
//
#include <hip/hip_runtime.h>
#include <hip/hip_bf16.h>

// ---------------------------------------------------------------------------
// Model dims (compile-time constants, match reference)
// ---------------------------------------------------------------------------
#define BATCH 2
#define LSEQ 256
#define CHAR_LEN 200
#define HDIM 256
#define NHEAD 8
#define DHEAD 32
#define FFDIM 1024
#define NLAYER 4
#define PE_ROWS 513          // 2*MAXLEN+1
#define LABELS 20
#define EDIM 50

typedef __bf16 bf16_t;
typedef __attribute__((ext_vector_type(16))) __bf16 v16bf;
typedef __attribute__((ext_vector_type(8)))  __bf16 v8bf;
typedef __attribute__((ext_vector_type(8)))  float  v8f;

__device__ __forceinline__ unsigned f2bfu(float x) {
  unsigned u = __builtin_bit_cast(unsigned, x);
  return (u + 0x7FFFu + ((u >> 16) & 1u)) >> 16;   // round-to-nearest-even
}
__device__ __forceinline__ unsigned pack2bf(float lo, float hi) {
  return f2bfu(lo) | (f2bfu(hi) << 16);
}
__device__ __forceinline__ bf16_t f2bf(float x) {
  return __builtin_bit_cast(bf16_t, (unsigned short)f2bfu(x));
}
__device__ __forceinline__ v16bf cat8(v8bf lo, v8bf hi) {
  return __builtin_shufflevector(lo, hi, 0, 1, 2, 3, 4, 5, 6, 7,
                                 8, 9, 10, 11, 12, 13, 14, 15);
}

// ---------------------------------------------------------------------------
// Generic bf16 WMMA GEMM:  C[M,N] = op(A[M,K] @ B[K,N] + bias), f32 in/out.
// Block = 256 threads = 8 wave32s arranged 4(M) x 2(N); each wave owns a
// 16x64 C strip (1 A frag reused over 4 B frags -> 4 v_wmma per K-step).
// Block tile: 64(M) x 128(N), K stepped by 32.
// LDS: aLds [M][K] bf16 (rows 16B aligned) -> A frag = 2x ds_load_b128.
//      bLds [N][K] bf16 (transposed at staging) -> B frag = 2x ds_load_b128.
// bT=1 means B is stored (N,K) row-major (then staging is K-major & packed).
// blockIdx.z batches via element offsets aBat/bBat/cBat.
// Requires K % 32 == 0 and N % 4 == 0.
// ---------------------------------------------------------------------------
__global__ __launch_bounds__(256) void gemm_bf16_wmma(
    const float* __restrict__ A, long aBat, int lda,
    const float* __restrict__ Bm, long bBat, int ldb, int bT,
    const float* __restrict__ bias,
    float* __restrict__ C, long cBat, int ldc,
    int M, int N, int K, int doRelu)
{
  __shared__ bf16_t aLds[64][40];    // 64 M rows x 32 K (+pad to 80B rows)
  __shared__ bf16_t bLds[128][40];   // 128 N rows x 32 K (+pad to 80B rows)

  const int z = blockIdx.z;
  A  += (long)z * aBat;
  Bm += (long)z * bBat;
  C  += (long)z * cBat;

  const int tid  = threadIdx.x;
  const int wave = tid >> 5;
  const int lane = tid & 31;
  const int msub = wave >> 1;        // 0..3
  const int nsub = wave & 1;         // 0..1
  const int m0 = blockIdx.y * 64;
  const int n0 = blockIdx.x * 128;

  v8f acc0 = {}, acc1 = {}, acc2 = {}, acc3 = {};

  for (int kk = 0; kk < K; kk += 32) {
    // ---- stage A tile 64x32 f32 -> bf16 [M][K] (2 float4 per thread)
#pragma unroll
    for (int it = 0; it < 2; ++it) {
      const int idx = tid + 256 * it;
      const int r  = idx >> 3;          // 0..63 (M)
      const int c4 = (idx & 7) * 4;     // 0,4,..,28 (K)
      const int gm = m0 + r;
      const int gmc = gm < M ? gm : (M - 1);
      float4 w = *(const float4*)(A + (long)gmc * lda + kk + c4);
      if (gm >= M) { w.x = 0.f; w.y = 0.f; w.z = 0.f; w.w = 0.f; }
      *(unsigned*)&aLds[r][c4]     = pack2bf(w.x, w.y);
      *(unsigned*)&aLds[r][c4 + 2] = pack2bf(w.z, w.w);
    }
    // ---- stage B tile -> bf16 [N][K] (transposed in LDS)
    if (!bT) {
      // B global is [K][N]: float4 along N, scatter 4 LDS rows
#pragma unroll
      for (int it = 0; it < 4; ++it) {
        const int idx = tid + 256 * it;
        const int r  = idx >> 5;           // 0..31 (K)
        const int c4 = (idx & 31) * 4;     // 0..124 (N)
        const int gn = n0 + c4;
        const int gnc = gn < N ? gn : (N - 4);
        float4 w = *(const float4*)(Bm + (long)(kk + r) * ldb + gnc);
        if (gn >= N) { w.x = 0.f; w.y = 0.f; w.z = 0.f; w.w = 0.f; }
        bLds[c4 + 0][r] = f2bf(w.x);
        bLds[c4 + 1][r] = f2bf(w.y);
        bLds[c4 + 2][r] = f2bf(w.z);
        bLds[c4 + 3][r] = f2bf(w.w);
      }
    } else {
      // B global is (N,K): float4 along K -> packed pair stores
#pragma unroll
      for (int it = 0; it < 4; ++it) {
        const int idx = tid + 256 * it;
        const int c  = idx >> 3;           // 0..127 (N)
        const int r4 = (idx & 7) * 4;      // 0,4,..,28 (K)
        const int gn = n0 + c;
        const int gnc = gn < N ? gn : (N - 1);
        float4 w = *(const float4*)(Bm + (long)gnc * ldb + kk + r4);
        if (gn >= N) { w.x = 0.f; w.y = 0.f; w.z = 0.f; w.w = 0.f; }
        *(unsigned*)&bLds[c][r4]     = pack2bf(w.x, w.y);
        *(unsigned*)&bLds[c][r4 + 2] = pack2bf(w.z, w.w);
      }
    }
    // prefetch next K-step tiles into cache while this step computes
    if (kk + 32 < K) {
      const int pr = tid >> 3;
      const int pm = (m0 + pr) < M ? (m0 + pr) : (M - 1);
      __builtin_prefetch(A + (long)pm * lda + kk + 32, 0, 1);
      if (!bT)
        __builtin_prefetch(Bm + (long)(kk + 32 + (tid >> 5)) * ldb + n0 + ((tid & 31) * 4), 0, 1);
    }
    __syncthreads();

    // ---- fragments (vector LDS loads) ----
    // A (16x32): lanes 0-15: M=lane, K 0-15; lanes 16-31: K 16-31.
    const int am  = msub * 16 + (lane & 15);
    const int khi = (lane >> 4) << 4;          // 0 or 16
    const bf16_t* ap = &aLds[am][khi];
    const v16bf af = cat8(*(const v8bf*)ap, *(const v8bf*)(ap + 8));
    // B (32x16): lane = N column; lanes 0-15 K=0..15, lanes 16-31 K=16..31.
    const int bk0 = (lane >> 4) << 4;          // 0 or 16
    const int bn  = nsub * 64 + (lane & 15);
    const bf16_t* bp0 = &bLds[bn][bk0];
    const bf16_t* bp1 = bp0 + 16 * 40;
    const bf16_t* bp2 = bp0 + 32 * 40;
    const bf16_t* bp3 = bp0 + 48 * 40;
    const v16bf bf0 = cat8(*(const v8bf*)bp0, *(const v8bf*)(bp0 + 8));
    const v16bf bf1 = cat8(*(const v8bf*)bp1, *(const v8bf*)(bp1 + 8));
    const v16bf bf2 = cat8(*(const v8bf*)bp2, *(const v8bf*)(bp2 + 8));
    const v16bf bf3 = cat8(*(const v8bf*)bp3, *(const v8bf*)(bp3 + 8));

    acc0 = __builtin_amdgcn_wmma_f32_16x16x32_bf16(false, af, false, bf0, (short)0, acc0, false, false);
    acc1 = __builtin_amdgcn_wmma_f32_16x16x32_bf16(false, af, false, bf1, (short)0, acc1, false, false);
    acc2 = __builtin_amdgcn_wmma_f32_16x16x32_bf16(false, af, false, bf2, (short)0, acc2, false, false);
    acc3 = __builtin_amdgcn_wmma_f32_16x16x32_bf16(false, af, false, bf3, (short)0, acc3, false, false);
    __syncthreads();
  }

  // ---- epilogue: VGPR r -> M = r (+8 for lanes 16-31) ----
  const int cmBase = m0 + msub * 16 + ((lane >> 4) << 3);
#pragma unroll
  for (int bi = 0; bi < 4; ++bi) {
    const v8f acc = bi == 0 ? acc0 : (bi == 1 ? acc1 : (bi == 2 ? acc2 : acc3));
    const int cn = n0 + nsub * 64 + bi * 16 + (lane & 15);
    if (cn < N) {
      const float bv = bias ? bias[cn] : 0.0f;
#pragma unroll
      for (int r = 0; r < 8; ++r) {
        const int cm = cmBase + r;
        if (cm < M) {
          float v = acc[r] + bv;
          if (doRelu) v = fmaxf(v, 0.0f);
          C[(long)cm * ldc + cn] = v;
        }
      }
    }
  }
}

// ---------------------------------------------------------------------------
// Embedding: x[b,i,:] = char_mask*(concat(raw,big)@cW + cb) + lex_mask*(raw@lW + lb)
// ---------------------------------------------------------------------------
__global__ __launch_bounds__(256) void embed_kernel(
    const int* __restrict__ lattice, const int* __restrict__ bigrams,
    const int* __restrict__ seq_len, const int* __restrict__ lex_num,
    const float* __restrict__ lat_tab, const float* __restrict__ big_tab,
    const float* __restrict__ cW, const float* __restrict__ cb,
    const float* __restrict__ lW, const float* __restrict__ lb,
    float* __restrict__ x)
{
  const int t = blockIdx.x;           // b*LSEQ + i
  const int b = t >> 8;
  const int i = t & 255;
  __shared__ float rc[2 * EDIM];      // raw_char = [raw(50) | big(50)]
  const int tid = threadIdx.x;
  if (tid < EDIM) {
    rc[tid] = lat_tab[(long)lattice[t] * EDIM + tid];
  } else if (tid < 2 * EDIM) {
    int k2 = tid - EDIM;
    rc[tid] = (i < CHAR_LEN)
                  ? big_tab[(long)bigrams[b * CHAR_LEN + i] * EDIM + k2]
                  : 0.0f;
  }
  __syncthreads();
  const int sl = seq_len[b];
  const bool cm = i < sl;
  const bool tm = i < sl + lex_num[b];
  const bool lm = tm && !cm;
  const int c = tid;
  float acc = 0.0f;
  if (cm) {
    float a = cb[c];
    for (int k = 0; k < 2 * EDIM; ++k) a += rc[k] * cW[k * HDIM + c];
    acc += a;
  }
  if (lm) {
    float a = lb[c];
    for (int k = 0; k < EDIM; ++k) a += rc[k] * lW[k * HDIM + c];
    acc += a;
  }
  x[(long)t * HDIM + c] = acc;
}

// ---------------------------------------------------------------------------
// qu = q + u[l], qv = q + vb[l]
// ---------------------------------------------------------------------------
__global__ __launch_bounds__(256) void quv_kernel(
    const float* __restrict__ q, const float* __restrict__ u,
    const float* __restrict__ vb, float* __restrict__ qu,
    float* __restrict__ qv)
{
  const int c = threadIdx.x;
  const long idx = (long)blockIdx.x * HDIM + c;
  const float qq = q[idx];
  qu[idx] = qq + u[c];
  qv[idx] = qq + vb[c];
}

// ---------------------------------------------------------------------------
// Fused attention per (b,i):
//   AC[j,h] = qu_i[h,:].k[b,j,h,:]
//   BD[j,h] = relu(peW0[dss]+peW1[dse]+peW2[des]+peW3[dee]+fb) . w_bd[b,i,h,:]
//   p = softmax_j(mask((AC+BD)*scale));  o[b,i,h,d] = sum_j p[j,h]*v[b,j,h,d]
// ---------------------------------------------------------------------------
__global__ __launch_bounds__(256) void attn_kernel(
    const float* __restrict__ qu, const float* __restrict__ k,
    const float* __restrict__ v, const float* __restrict__ w_bd,
    const float* __restrict__ peW, const float* __restrict__ fusion_b,
    const int* __restrict__ pos_s, const int* __restrict__ pos_e,
    const int* __restrict__ seq_len, const int* __restrict__ lex_num,
    float* __restrict__ o)
{
  const int t = blockIdx.x;            // b*LSEQ + i
  const int b = t >> 8;
  const int tid = threadIdx.x;
  const int wave = tid >> 5;
  const int lane = tid & 31;

  __shared__ float quS[HDIM];
  __shared__ float wS[NHEAD][HDIM];
  __shared__ float sS[NHEAD][LSEQ];
  __shared__ int psS[LSEQ];
  __shared__ int peS[LSEQ];

  quS[tid] = qu[(long)t * HDIM + tid];
  psS[tid] = pos_s[b * LSEQ + tid];
  peS[tid] = pos_e[b * LSEQ + tid];
#pragma unroll
  for (int h = 0; h < NHEAD; ++h)
    wS[h][tid] = w_bd[((long)t * NHEAD + h) * HDIM + tid];
  const int total = seq_len[b] + lex_num[b];
  const int ps_i = pos_s[t];
  const int pe_i = pos_e[t];
  __syncthreads();

  // ---- AC: thread j, loop heads ----
  {
    const int j = tid;
    const float* kj = k + ((long)(b * LSEQ + j)) * HDIM;
#pragma unroll
    for (int h = 0; h < NHEAD; ++h) {
      float a = 0.0f;
      for (int d = 0; d < DHEAD; ++d) a += quS[h * DHEAD + d] * kj[h * DHEAD + d];
      sS[h][j] = a;
    }
  }
  __syncthreads();

  // ---- BD: each wave handles 32 j values; lanes split channels ----
  for (int jj = 0; jj < 32; ++jj) {
    const int j = wave * 32 + jj;
    const long dss = (long)(ps_i - psS[j] + 256) * HDIM;
    const long dse = (long)(ps_i - peS[j] + 256) * HDIM;
    const long des = (long)(pe_i - psS[j] + 256) * HDIM;
    const long dee = (long)(pe_i - peS[j] + 256) * HDIM;
    const float* p0 = peW + dss;
    const float* p1 = peW + (long)PE_ROWS * HDIM + dse;
    const float* p2 = peW + 2L * PE_ROWS * HDIM + des;
    const float* p3 = peW + 3L * PE_ROWS * HDIM + dee;
    if (jj + 1 < 32) {     // prefetch next j's gather rows (L2-resident tables)
      const int jn = j + 1;
      __builtin_prefetch(peW + (long)(ps_i - psS[jn] + 256) * HDIM + lane * 8, 0, 1);
      __builtin_prefetch(peW + (long)PE_ROWS * HDIM + (long)(ps_i - peS[jn] + 256) * HDIM + lane * 8, 0, 1);
      __builtin_prefetch(peW + 2L * PE_ROWS * HDIM + (long)(pe_i - psS[jn] + 256) * HDIM + lane * 8, 0, 1);
      __builtin_prefetch(peW + 3L * PE_ROWS * HDIM + (long)(pe_i - peS[jn] + 256) * HDIM + lane * 8, 0, 1);
    }
    float part[NHEAD] = {};
#pragma unroll
    for (int e = 0; e < 8; ++e) {
      const int c = e * 32 + lane;               // coalesced across lanes
      float r = p0[c] + p1[c] + p2[c] + p3[c] + fusion_b[c];
      r = fmaxf(r, 0.0f);
#pragma unroll
      for (int h = 0; h < NHEAD; ++h) part[h] += r * wS[h][c];
    }
#pragma unroll
    for (int h = 0; h < NHEAD; ++h)
      for (int off = 16; off > 0; off >>= 1)
        part[h] += __shfl_xor(part[h], off, 32);
    if (lane == 0) {
#pragma unroll
      for (int h = 0; h < NHEAD; ++h) sS[h][j] += part[h];
    }
  }
  __syncthreads();

  // ---- masked softmax over j; wave w owns head w ----
  {
    const int h = wave;
    const float scale = 0.1767766952966369f;     // 1/sqrt(32)
    float mx = -3.0e38f;
    for (int j = lane; j < LSEQ; j += 32) {
      float sc = (j < total) ? sS[h][j] * scale : -1.0e15f;
      sS[h][j] = sc;
      mx = fmaxf(mx, sc);
    }
    for (int off = 16; off > 0; off >>= 1)
      mx = fmaxf(mx, __shfl_xor(mx, off, 32));
    float sum = 0.0f;
    for (int j = lane; j < LSEQ; j += 32) {
      float e = __expf(sS[h][j] - mx);
      sS[h][j] = e;
      sum += e;
    }
    for (int off = 16; off > 0; off >>= 1)
      sum += __shfl_xor(sum, off, 32);
    const float inv = 1.0f / sum;
    for (int j = lane; j < LSEQ; j += 32) sS[h][j] *= inv;
  }
  __syncthreads();

  // ---- o = p @ v : thread = (h,d) channel ----
  {
    const int h = tid >> 5;
    const int d = tid & 31;
    float acc = 0.0f;
    const float* vp = v + (long)b * LSEQ * HDIM + h * DHEAD + d;
    for (int j = 0; j < LSEQ; ++j) acc += sS[h][j] * vp[(long)j * HDIM];
    o[(long)t * HDIM + h * DHEAD + d] = acc;
  }
}

// ---------------------------------------------------------------------------
// LayerNorm of (in + in): block per token row.
// ---------------------------------------------------------------------------
__global__ __launch_bounds__(256) void ln2x_kernel(
    const float* __restrict__ in, const float* __restrict__ g,
    const float* __restrict__ bb, float* __restrict__ out)
{
  const int row = blockIdx.x;
  const int c = threadIdx.x;
  const float y = 2.0f * in[(long)row * HDIM + c];
  float s1 = y, s2 = y * y;
  for (int off = 16; off > 0; off >>= 1) {
    s1 += __shfl_xor(s1, off, 32);
    s2 += __shfl_xor(s2, off, 32);
  }
  __shared__ float w1[8], w2[8];
  const int wave = c >> 5, lane = c & 31;
  if (lane == 0) { w1[wave] = s1; w2[wave] = s2; }
  __syncthreads();
  if (c == 0) {
    float a = 0.0f, b2 = 0.0f;
    for (int i = 0; i < 8; ++i) { a += w1[i]; b2 += w2[i]; }
    w1[0] = a; w2[0] = b2;
  }
  __syncthreads();
  const float mean = w1[0] * (1.0f / HDIM);
  const float var = w2[0] * (1.0f / HDIM) - mean * mean;
  const float inv = rsqrtf(var + 1e-5f);
  out[(long)row * HDIM + c] = (y - mean) * inv * g[c] + bb[c];
}

// ---------------------------------------------------------------------------
// Final classifier: enc = x[:, :CHAR_LEN] @ out_W + out_b
// ---------------------------------------------------------------------------
__global__ void out_kernel(const float* __restrict__ x,
                           const float* __restrict__ W,
                           const float* __restrict__ bias,
                           float* __restrict__ out)
{
  const int t = blockIdx.x;            // 0 .. B*CHAR_LEN-1
  const int b = t / CHAR_LEN;
  const int i = t % CHAR_LEN;
  const float* xr = x + ((long)(b * LSEQ + i)) * HDIM;
  const int c = threadIdx.x;
  if (c < LABELS) {
    float a = bias[c];
    for (int k = 0; k < HDIM; ++k) a += xr[k] * W[k * LABELS + c];
    out[(long)t * LABELS + c] = a;
  }
}

// ---------------------------------------------------------------------------
// Host orchestration
// ---------------------------------------------------------------------------
static inline void launch_gemm(hipStream_t s,
                               const float* A, long aBat, int lda,
                               const float* Bm, long bBat, int ldb, int bT,
                               const float* bias,
                               float* C, long cBat, int ldc,
                               int M, int N, int K, int relu, int batch)
{
  dim3 grid((N + 127) / 128, (M + 63) / 64, batch);
  gemm_bf16_wmma<<<grid, 256, 0, s>>>(A, aBat, lda, Bm, bBat, ldb, bT,
                                      bias, C, cBat, ldc, M, N, K, relu);
}

extern "C" void kernel_launch(void* const* d_in, const int* in_sizes, int n_in,
                              void* d_out, int out_size, void* d_ws, size_t ws_size,
                              hipStream_t stream) {
  (void)in_sizes; (void)n_in; (void)out_size; (void)ws_size;

  const int*   lattice   = (const int*)d_in[0];
  const int*   bigrams   = (const int*)d_in[1];
  const int*   seq_len   = (const int*)d_in[2];
  const int*   lex_num   = (const int*)d_in[3];
  const int*   pos_s     = (const int*)d_in[4];
  const int*   pos_e     = (const int*)d_in[5];
  const float* lat_tab   = (const float*)d_in[6];
  const float* big_tab   = (const float*)d_in[7];
  const float* cW        = (const float*)d_in[8];
  const float* cb        = (const float*)d_in[9];
  const float* lW        = (const float*)d_in[10];
  const float* lb        = (const float*)d_in[11];
  const float* pe        = (const float*)d_in[12];
  const float* fusion_W  = (const float*)d_in[13];
  const float* fusion_b  = (const float*)d_in[14];
  const float* Wq        = (const float*)d_in[15];
  const float* bq        = (const float*)d_in[16];
  const float* Wk        = (const float*)d_in[17];
  const float* bk        = (const float*)d_in[18];
  const float* Wv        = (const float*)d_in[19];
  const float* bv        = (const float*)d_in[20];
  const float* Wr        = (const float*)d_in[21];
  /* br (d_in[22]) contributes a j-constant to pre-softmax scores -> cancels */
  const float* u_tab     = (const float*)d_in[23];
  const float* vb_tab    = (const float*)d_in[24];
  const float* Wo        = (const float*)d_in[25];
  const float* bo        = (const float*)d_in[26];
  const float* ln1_g     = (const float*)d_in[27];
  const float* ln1_b     = (const float*)d_in[28];
  const float* W1        = (const float*)d_in[29];
  const float* b1        = (const float*)d_in[30];
  const float* W2        = (const float*)d_in[31];
  const float* b2        = (const float*)d_in[32];
  const float* ln2_g     = (const float*)d_in[33];
  const float* ln2_b     = (const float*)d_in[34];
  const float* out_W     = (const float*)d_in[35];
  const float* out_b     = (const float*)d_in[36];
  float* out = (float*)d_out;

  // workspace layout (floats)
  const long TOK = (long)BATCH * LSEQ;           // 512
  const long XSZ = TOK * HDIM;                   // 131072
  const long PEWSZ = (long)PE_ROWS * HDIM;       // 131328 per table
  float* ws   = (float*)d_ws;
  float* x    = ws;                 ws += XSZ;
  float* peW  = ws;                 ws += 4 * PEWSZ;
  float* q    = ws;                 ws += XSZ;
  float* kbuf = ws;                 ws += XSZ;
  float* vbuf = ws;                 ws += XSZ;
  float* qu   = ws;                 ws += XSZ;
  float* qv   = ws;                 ws += XSZ;
  float* wbd  = ws;                 ws += TOK * NHEAD * HDIM;  // 1M floats
  float* ao   = ws;                 ws += XSZ;
  float* oprj = ws;                 ws += XSZ;
  float* xo   = ws;                 ws += XSZ;
  float* h1   = ws;                 ws += TOK * FFDIM;
  float* h2   = ws;                 ws += XSZ;

  // 1) embeddings -> x
  embed_kernel<<<dim3((unsigned)TOK), 256, 0, stream>>>(
      lattice, bigrams, seq_len, lex_num, lat_tab, big_tab,
      cW, cb, lW, lb, x);

  // 2) peW[t] = pe @ fusion_W[t*256:(t+1)*256, :]   (z-batched over 4 chunks)
  launch_gemm(stream, pe, 0, HDIM,
              fusion_W, (long)HDIM * HDIM, HDIM, 0, nullptr,
              peW, PEWSZ, HDIM, PE_ROWS, HDIM, HDIM, 0, 4);

  for (int l = 0; l < NLAYER; ++l) {
    const long wOff = (long)l * HDIM * HDIM;
    const long bOff = (long)l * HDIM;

    // Q, K, V projections (WMMA)
    launch_gemm(stream, x, 0, HDIM, Wq + wOff, 0, HDIM, 0, bq + bOff,
                q, 0, HDIM, (int)TOK, HDIM, HDIM, 0, 1);
    launch_gemm(stream, x, 0, HDIM, Wk + wOff, 0, HDIM, 0, bk + bOff,
                kbuf, 0, HDIM, (int)TOK, HDIM, HDIM, 0, 1);
    launch_gemm(stream, x, 0, HDIM, Wv + wOff, 0, HDIM, 0, bv + bOff,
                vbuf, 0, HDIM, (int)TOK, HDIM, HDIM, 0, 1);

    // qu = q + u[l], qv = q + vb[l]
    quv_kernel<<<dim3((unsigned)TOK), 256, 0, stream>>>(
        q, u_tab + bOff, vb_tab + bOff, qu, qv);

    // w_bd[b,i,h,c] = sum_d qv[b,i,h*32+d] * Wr[l][c, h*32+d]
    // batched over h: A offset 32/h, B^T offset 32/h, C offset 256/h
    launch_gemm(stream, qv, 32, HDIM,
                Wr + wOff, 32, HDIM, 1, nullptr,
                wbd, HDIM, NHEAD * HDIM,
                (int)TOK, HDIM, DHEAD, 0, NHEAD);

    // fused attention
    attn_kernel<<<dim3((unsigned)TOK), 256, 0, stream>>>(
        qu, kbuf, vbuf, wbd, peW, fusion_b,
        pos_s, pos_e, seq_len, lex_num, ao);

    // output projection + ln1(o+o)
    launch_gemm(stream, ao, 0, HDIM, Wo + wOff, 0, HDIM, 0, bo + bOff,
                oprj, 0, HDIM, (int)TOK, HDIM, HDIM, 0, 1);
    ln2x_kernel<<<dim3((unsigned)TOK), 256, 0, stream>>>(
        oprj, ln1_g + bOff, ln1_b + bOff, xo);

    // FFN
    launch_gemm(stream, xo, 0, HDIM,
                W1 + (long)l * HDIM * FFDIM, 0, FFDIM, 0, b1 + (long)l * FFDIM,
                h1, 0, FFDIM, (int)TOK, FFDIM, HDIM, 1, 1);
    launch_gemm(stream, h1, 0, FFDIM,
                W2 + (long)l * FFDIM * HDIM, 0, HDIM, 0, b2 + bOff,
                h2, 0, HDIM, (int)TOK, HDIM, FFDIM, 0, 1);

    // x = ln2(h+h)
    ln2x_kernel<<<dim3((unsigned)TOK), 256, 0, stream>>>(
        h2, ln2_g + bOff, ln2_b + bOff, x);
  }

  // classifier on first CHAR_LEN tokens of each batch
  out_kernel<<<dim3(BATCH * CHAR_LEN), 32, 0, stream>>>(x, out_W, out_b, out);
}
